// HDAttention_50139448214028
// MI455X (gfx1250) — matrix operand order repaired and checked by reading, not compiled, not run on hardware
//
#include <hip/hip_runtime.h>
#include <hip/hip_bf16.h>

// ---------------------------------------------------------------------------
// Problem constants (from reference): B=1, LI=2048, LT=384, DM=2560, H=20, DH=128
// ---------------------------------------------------------------------------
static constexpr int kLI  = 2048;
static constexpr int kLT  = 384;
static constexpr int kL   = 2432;   // LI + LT
static constexpr int kDM  = 2560;
static constexpr int kH   = 20;
static constexpr int kDH  = 128;

typedef __attribute__((ext_vector_type(16))) __bf16 v16bf;
typedef __attribute__((ext_vector_type(8)))  __bf16 v8bf;
typedef __attribute__((ext_vector_type(8)))  float  v8f;

// Build a 16-element bf16 A/B fragment from two 16-byte chunks.
__device__ __forceinline__ v16bf frag16(const __bf16* p0, const __bf16* p1) {
    v8bf a = *(const v8bf*)p0;
    v8bf b = *(const v8bf*)p1;
    return __builtin_shufflevector(a, b, 0,1,2,3,4,5,6,7,8,9,10,11,12,13,14,15);
}

__device__ __forceinline__ v8f wmma_bf16(v16bf a, v16bf b, v8f c) {
    // (neg_a, A, neg_b, B, c_mod, C, reuse_a, reuse_b)
    return __builtin_amdgcn_wmma_f32_16x16x32_bf16(false, a, false, b, (short)0, c,
                                                   false, false);
}

// CDNA5 async global -> LDS copy (VGPR-bypassing, tracked by ASYNCcnt).
// vdst VGPR holds the LDS byte address, vaddr the 64-bit global address.
__device__ __forceinline__ void async_load_b128(unsigned lds_addr, const void* gaddr) {
    asm volatile("global_load_async_to_lds_b128 %0, %1, off"
                 :: "v"(lds_addr), "v"(gaddr)
                 : "memory");
}

__device__ __forceinline__ void wait_asynccnt0() {
    asm volatile("s_wait_asynccnt 0x0" ::: "memory");
}

// ---------------------------------------------------------------------------
// f32 -> bf16 downconvert (grid-stride, 4 elements per thread)
// ---------------------------------------------------------------------------
__global__ __launch_bounds__(256)
void f32_to_bf16_kernel(const float* __restrict__ src, __bf16* __restrict__ dst,
                        size_t n) {
    size_t i = ((size_t)blockIdx.x * 256 + threadIdx.x) * 4;
    size_t stride = (size_t)gridDim.x * 256 * 4;
    for (; i < n; i += stride) {
        float4 v = *(const float4*)(src + i);
        dst[i + 0] = (__bf16)v.x;
        dst[i + 1] = (__bf16)v.y;
        dst[i + 2] = (__bf16)v.z;
        dst[i + 3] = (__bf16)v.w;
    }
}

// ---------------------------------------------------------------------------
// GEMM: C[M,N] (f32) = A[M,K] (bf16, row major) x W[N,K]^T (bf16, row major) + bias[N]
// Block tile 128x128, K-stage 64. Double-buffered LDS, tiles staged with
// global_load_async_to_lds_b128 (ASYNCcnt), one barrier per K-step.
// 8 waves: each wave computes 2 (M) x 4 (N) WMMA tiles.
// M, N multiples of 128; K multiple of 64 (true for all our shapes).
// ---------------------------------------------------------------------------
__global__ __launch_bounds__(256)
void gemm_bf16_kernel(const __bf16* __restrict__ A, const __bf16* __restrict__ W,
                      const float* __restrict__ bias, float* __restrict__ C,
                      int M, int N, int K) {
    __shared__ __bf16 ldsA[2][128 * 72];   // 64 halfs + 8 pad per row
    __shared__ __bf16 ldsB[2][128 * 72];

    const int tid   = threadIdx.x;
    const int lane  = tid & 31;
    const int wave  = tid >> 5;
    const int waveM = wave & 3;     // 4 slots of 32 rows
    const int waveN = wave >> 2;    // 2 slots of 64 cols
    const int l16   = lane & 15;
    const int lhi   = lane >> 4;

    const int rowBase = blockIdx.y * 128;
    const int colBase = blockIdx.x * 128;

    // Per-thread staging geometry: 4 x 16B chunks per matrix per stage.
    const int r0  = tid >> 1;               // rows r0, r0+... via chunk id below
    (void)r0;

    v8f zero = {};
    v8f acc[2][4];
    #pragma unroll
    for (int i = 0; i < 2; ++i)
        #pragma unroll
        for (int j = 0; j < 4; ++j) acc[i][j] = zero;

    // Issue async stage of tile kb into LDS buffer `buf`.
    auto stage = [&](int buf, int kb) {
        unsigned baseA = (unsigned)(size_t)&ldsA[buf][0];
        unsigned baseB = (unsigned)(size_t)&ldsB[buf][0];
        #pragma unroll
        for (int i = 0; i < 4; ++i) {
            int c  = tid + 256 * i;     // 0..1023 chunk id
            int r  = c >> 3;            // 0..127 row
            int c8 = (c & 7) * 8;       // 0..56 col (halfs)
            unsigned loff = (unsigned)(r * 72 + c8) * 2u;
            async_load_b128(baseA + loff, &A[(size_t)(rowBase + r) * K + kb + c8]);
            async_load_b128(baseB + loff, &W[(size_t)(colBase + r) * K + kb + c8]);
        }
    };

    stage(0, 0);                            // prologue prefetch

    int cur = 0;
    for (int kb = 0; kb < K; kb += 64) {
        wait_asynccnt0();                   // my async copies for `cur` landed
        __syncthreads();                    // everyone's landed; prev compute done
        if (kb + 64 < K) stage(cur ^ 1, kb + 64);   // prefetch next tile

        #pragma unroll
        for (int kk = 0; kk < 64; kk += 32) {
            v16bf af[2];
            #pragma unroll
            for (int mt = 0; mt < 2; ++mt) {
                int r = waveM * 32 + mt * 16 + l16;
                const __bf16* p = &ldsA[cur][r * 72 + kk + lhi * 8];
                af[mt] = frag16(p, p + 16);     // A layout: {K 0-7,16-23} / {8-15,24-31}
            }
            #pragma unroll
            for (int nt = 0; nt < 4; ++nt) {
                int cc = waveN * 64 + nt * 16 + l16;
                const __bf16* p = &ldsB[cur][cc * 72 + kk + lhi * 8];
                v16bf bf = frag16(p, p + 16);   // B layout: {K 0-7,16-23} / {8-15,24-31}
                #pragma unroll
                for (int mt = 0; mt < 2; ++mt)
                    acc[mt][nt] = wmma_bf16(af[mt], bf, acc[mt][nt]);
            }
        }
        cur ^= 1;
    }

    // Epilogue: C layout => VGPR r holds row (lhi*8 + r), column l16 within tile
    #pragma unroll
    for (int nt = 0; nt < 4; ++nt) {
        int col = colBase + waveN * 64 + nt * 16 + l16;
        float bv = bias[col];
        #pragma unroll
        for (int mt = 0; mt < 2; ++mt) {
            int row0 = rowBase + waveM * 32 + mt * 16 + lhi * 8;
            #pragma unroll
            for (int r = 0; r < 8; ++r)
                C[(size_t)(row0 + r) * N + col] = acc[mt][nt][r] + bv;
        }
    }
}

// ---------------------------------------------------------------------------
// RMS over full DM, multiply by norm weight (img vs txt rows), optional k-mask,
// apply RoPE, pack bf16 as [H][L][DH]. One block (256 threads) per token.
// ---------------------------------------------------------------------------
__global__ __launch_bounds__(256)
void rms_rope_pack_kernel(const float* __restrict__ y,
                          const float* __restrict__ nw_img,
                          const float* __restrict__ nw_txt,
                          const float* __restrict__ rope,
                          const float* __restrict__ mask,
                          int useMask,
                          __bf16* __restrict__ out) {
    const int t   = blockIdx.x;             // global token position 0..L-1
    const int tid = threadIdx.x;
    const float* yrow = y + (size_t)t * kDM;

    float ss = 0.0f;
    #pragma unroll
    for (int i = 0; i < 10; ++i) {          // 2560 = 256*10
        float v = yrow[tid + 256 * i];
        ss += v * v;
    }
    #pragma unroll
    for (int m = 1; m < 32; m <<= 1) ss += __shfl_xor(ss, m, 32);

    __shared__ float red[8];
    __shared__ float s_rms, s_mask;
    if ((tid & 31) == 0) red[tid >> 5] = ss;
    __syncthreads();
    if (tid == 0) {
        float tot = 0.0f;
        #pragma unroll
        for (int i = 0; i < 8; ++i) tot += red[i];
        s_rms  = rsqrtf(tot / (float)kDM + 1e-5f);
        s_mask = (useMask && (t < kLI)) ? mask[t] : 1.0f;
    }
    __syncthreads();

    const float rms = s_rms;
    const float mk  = s_mask;
    const float* nw = (t < kLI) ? nw_img : nw_txt;

    // 20 heads * 64 pairs = 1280 pairs = 256*5
    #pragma unroll
    for (int j = 0; j < 5; ++j) {
        int p  = tid + 256 * j;
        int h  = p >> 6;
        int pr = p & 63;
        int cb = h * kDH + pr * 2;
        float x0 = yrow[cb]     * rms * nw[cb]     * mk;
        float x1 = yrow[cb + 1] * rms * nw[cb + 1] * mk;
        // rope flat: [t][pr][j(out)][i(src)] -> t*256 + pr*4 + j*2 + i
        const float* rp = rope + (size_t)t * 256 + pr * 4;
        float o0 = rp[0] * x0 + rp[1] * x1;
        float o1 = rp[2] * x0 + rp[3] * x1;
        __bf16* orow = out + ((size_t)h * kL + t) * kDH + pr * 2;
        orow[0] = (__bf16)o0;
        orow[1] = (__bf16)o1;
    }
}

// ---------------------------------------------------------------------------
// Pack V transposed: yv f32 [L][DM] -> vt bf16 [H][DH][L]
// ---------------------------------------------------------------------------
__global__ __launch_bounds__(256)
void v_pack_kernel(const float* __restrict__ yv, __bf16* __restrict__ vt) {
    size_t idx    = (size_t)blockIdx.x * 256 + threadIdx.x;
    size_t total  = (size_t)kL * kDM;
    size_t stride = (size_t)gridDim.x * 256;
    for (; idx < total; idx += stride) {
        int t = (int)(idx / kDM);
        int c = (int)(idx % kDM);
        int h = c >> 7;
        int d = c & 127;
        vt[(size_t)(h * kDH + d) * kL + t] = (__bf16)yv[idx];
    }
}

// ---------------------------------------------------------------------------
// Flash attention. q,k: bf16 [H][L][DH]; vt: bf16 [H][DH][L]; attn out bf16 [L][DM].
// Grid: (L/128, H). 8 waves/block, each wave owns 16 query rows.
// Key blocks of 64 (L = 2432 = 38*64 exactly). K/V stream from L2 (25 MB total,
// fits the 192 MB L2 easily).
// ---------------------------------------------------------------------------
__global__ __launch_bounds__(256)
void flash_attn_kernel(const __bf16* __restrict__ q,
                       const __bf16* __restrict__ k,
                       const __bf16* __restrict__ vt,
                       __bf16* __restrict__ attn) {
    __shared__ __bf16 ldsP[8][16 * 72];     // per-wave P staging (16x64, padded)

    const int tid   = threadIdx.x;
    const int lane  = tid & 31;
    const int wave  = tid >> 5;
    const int l16   = lane & 15;
    const int lhi   = lane >> 4;
    const int h     = blockIdx.y;
    const int qBase = blockIdx.x * 128 + wave * 16;

    const __bf16* qh = q  + (size_t)h * kL * kDH;
    const __bf16* kh = k  + (size_t)h * kL * kDH;
    const __bf16* vh = vt + (size_t)h * kDH * kL;

    // Q fragments for all 4 K-steps of DH=128 (loaded once)
    v16bf qf[4];
    {
        const __bf16* qr = qh + (size_t)(qBase + l16) * kDH;
        #pragma unroll
        for (int ks = 0; ks < 4; ++ks) {
            const __bf16* p = qr + ks * 32 + lhi * 8;
            qf[ks] = frag16(p, p + 16);
        }
    }

    v8f zero = {};
    v8f o[8];
    #pragma unroll
    for (int i = 0; i < 8; ++i) o[i] = zero;
    float mrow[8], lrow[8];
    #pragma unroll
    for (int r = 0; r < 8; ++r) { mrow[r] = -1e30f; lrow[r] = 0.0f; }

    const float scale = 0.08838834764831845f;   // 1/sqrt(128)

    for (int kb = 0; kb < kL; kb += 64) {
        // S = Q * K^T  (16 x 64 per wave)
        v8f s[4];
        #pragma unroll
        for (int i = 0; i < 4; ++i) s[i] = zero;
        #pragma unroll
        for (int nt = 0; nt < 4; ++nt) {
            const __bf16* kr = kh + (size_t)(kb + nt * 16 + l16) * kDH;
            #pragma unroll
            for (int ks = 0; ks < 4; ++ks) {
                const __bf16* p = kr + ks * 32 + lhi * 8;
                v16bf kf = frag16(p, p + 16);
                s[nt] = wmma_bf16(qf[ks], kf, s[nt]);
            }
        }
        #pragma unroll
        for (int nt = 0; nt < 4; ++nt)
            #pragma unroll
            for (int r = 0; r < 8; ++r) s[nt][r] *= scale;

        // Online softmax: rows live across 16 lanes (same lhi group)
        float alpha[8];
        #pragma unroll
        for (int r = 0; r < 8; ++r) {
            float v = fmaxf(fmaxf(s[0][r], s[1][r]), fmaxf(s[2][r], s[3][r]));
            v = fmaxf(v, __shfl_xor(v, 1, 32));
            v = fmaxf(v, __shfl_xor(v, 2, 32));
            v = fmaxf(v, __shfl_xor(v, 4, 32));
            v = fmaxf(v, __shfl_xor(v, 8, 32));
            float mn = fmaxf(mrow[r], v);
            alpha[r] = __expf(mrow[r] - mn);
            mrow[r]  = mn;
        }
        #pragma unroll
        for (int r = 0; r < 8; ++r) {
            float sum = 0.0f;
            #pragma unroll
            for (int nt = 0; nt < 4; ++nt) {
                float p = __expf(s[nt][r] - mrow[r]);
                s[nt][r] = p;
                sum += p;
            }
            sum += __shfl_xor(sum, 1, 32);
            sum += __shfl_xor(sum, 2, 32);
            sum += __shfl_xor(sum, 4, 32);
            sum += __shfl_xor(sum, 8, 32);
            lrow[r] = lrow[r] * alpha[r] + sum;
        }
        #pragma unroll
        for (int n8 = 0; n8 < 8; ++n8)
            #pragma unroll
            for (int r = 0; r < 8; ++r) o[n8][r] *= alpha[r];

        // Transpose P (C layout) -> A layout via per-wave LDS staging
        #pragma unroll
        for (int nt = 0; nt < 4; ++nt) {
            int col = nt * 16 + l16;
            #pragma unroll
            for (int r = 0; r < 8; ++r) {
                int rowm = lhi * 8 + r;
                ldsP[wave][rowm * 72 + col] = (__bf16)s[nt][r];
            }
        }
        // O += P * V   (K-dim = 64 keys, 2 steps)
        #pragma unroll
        for (int ks2 = 0; ks2 < 2; ++ks2) {
            const __bf16* pp = &ldsP[wave][l16 * 72 + ks2 * 32 + lhi * 8];
            v16bf pf = frag16(pp, pp + 16);
            #pragma unroll
            for (int n8 = 0; n8 < 8; ++n8) {
                const __bf16* vp = vh + (size_t)(n8 * 16 + l16) * kL
                                      + kb + ks2 * 32 + lhi * 8;
                v16bf vf = frag16(vp, vp + 16);
                o[n8] = wmma_bf16(pf, vf, o[n8]);
            }
        }
    }

    // Finalize: O / l, pack to attn[L][DM] bf16
    float rl[8];
    #pragma unroll
    for (int r = 0; r < 8; ++r) rl[r] = 1.0f / lrow[r];
    #pragma unroll
    for (int n8 = 0; n8 < 8; ++n8) {
        int d = n8 * 16 + l16;
        #pragma unroll
        for (int r = 0; r < 8; ++r) {
            int row = qBase + lhi * 8 + r;
            attn[(size_t)row * kDM + h * kDH + d] = (__bf16)(o[n8][r] * rl[r]);
        }
    }
}

// ---------------------------------------------------------------------------
// Host launcher
// ---------------------------------------------------------------------------
extern "C" void kernel_launch(void* const* d_in, const int* in_sizes, int n_in,
                              void* d_out, int out_size, void* d_ws, size_t ws_size,
                              hipStream_t stream) {
    (void)in_sizes; (void)n_in; (void)out_size; (void)ws_size;

    const float* img  = (const float*)d_in[0];
    const float* txt  = (const float*)d_in[1];
    const float* rope = (const float*)d_in[2];
    const float* mask = (const float*)d_in[3];
    // dict order: Wq,bq, Wk,bk, Wv,bv, Wo,bo, Wq_t,bq_t, Wk_t,bk_t, Wv_t,bv_t, Wo_t,bo_t, qn,kn,qn_t,kn_t
    const float* Wf[8] = {
        (const float*)d_in[4],  (const float*)d_in[6],  (const float*)d_in[8],
        (const float*)d_in[10], (const float*)d_in[12], (const float*)d_in[14],
        (const float*)d_in[16], (const float*)d_in[18]
    }; // order: Wq, Wk, Wv, Wo, Wq_t, Wk_t, Wv_t, Wo_t
    const float* bq   = (const float*)d_in[5];
    const float* bk   = (const float*)d_in[7];
    const float* bv   = (const float*)d_in[9];
    const float* bo   = (const float*)d_in[11];
    const float* bq_t = (const float*)d_in[13];
    const float* bk_t = (const float*)d_in[15];
    const float* bv_t = (const float*)d_in[17];
    const float* bo_t = (const float*)d_in[19];
    const float* qn   = (const float*)d_in[20];
    const float* kn   = (const float*)d_in[21];
    const float* qn_t = (const float*)d_in[22];
    const float* kn_t = (const float*)d_in[23];

    // Workspace carve-out
    char* ws = (char*)d_ws;
    size_t off = 0;
    auto carve = [&](size_t bytes) -> char* {
        char* p = ws + off;
        off += (bytes + 255) & ~(size_t)255;
        return p;
    };
    __bf16* Acat = (__bf16*)carve((size_t)kL * kDM * 2);
    __bf16* Wbf  = (__bf16*)carve((size_t)8 * kDM * kDM * 2);
    float*  yq   = (float*) carve((size_t)kL * kDM * 4);
    float*  yk   = (float*) carve((size_t)kL * kDM * 4);
    float*  yv   = (float*) carve((size_t)kL * kDM * 4);
    __bf16* qbf  = (__bf16*)carve((size_t)kH * kL * kDH * 2);
    __bf16* kbf  = (__bf16*)carve((size_t)kH * kL * kDH * 2);
    __bf16* vtbf = (__bf16*)carve((size_t)kH * kDH * kL * 2);
    __bf16* attn = (__bf16*)carve((size_t)kL * kDM * 2);

    // 1) downconvert activations + weights to bf16
    {
        size_t n = (size_t)kLI * kDM;
        f32_to_bf16_kernel<<<(unsigned)((n / 4 + 255) / 256), 256, 0, stream>>>(img, Acat, n);
        size_t n2 = (size_t)kLT * kDM;
        f32_to_bf16_kernel<<<(unsigned)((n2 / 4 + 255) / 256), 256, 0, stream>>>(
            txt, Acat + (size_t)kLI * kDM, n2);
    }
    for (int i = 0; i < 8; ++i) {
        size_t n = (size_t)kDM * kDM;
        f32_to_bf16_kernel<<<(unsigned)((n / 4 + 255) / 256), 256, 0, stream>>>(
            Wf[i], Wbf + (size_t)i * kDM * kDM, n);
    }

    const __bf16* Aimg = Acat;
    const __bf16* Atxt = Acat + (size_t)kLI * kDM;
    dim3 blk(256);
    dim3 gridImg(kDM / 128, kLI / 128);   // (20, 16)
    dim3 gridTxt(kDM / 128, kLT / 128);   // (20, 3)

    // 2) QKV projections (f32 pre-norm outputs)
    gemm_bf16_kernel<<<gridImg, blk, 0, stream>>>(Aimg, Wbf + (size_t)0 * kDM * kDM, bq,   yq,                       kLI, kDM, kDM);
    gemm_bf16_kernel<<<gridTxt, blk, 0, stream>>>(Atxt, Wbf + (size_t)4 * kDM * kDM, bq_t, yq + (size_t)kLI * kDM,  kLT, kDM, kDM);
    gemm_bf16_kernel<<<gridImg, blk, 0, stream>>>(Aimg, Wbf + (size_t)1 * kDM * kDM, bk,   yk,                       kLI, kDM, kDM);
    gemm_bf16_kernel<<<gridTxt, blk, 0, stream>>>(Atxt, Wbf + (size_t)5 * kDM * kDM, bk_t, yk + (size_t)kLI * kDM,  kLT, kDM, kDM);
    gemm_bf16_kernel<<<gridImg, blk, 0, stream>>>(Aimg, Wbf + (size_t)2 * kDM * kDM, bv,   yv,                       kLI, kDM, kDM);
    gemm_bf16_kernel<<<gridTxt, blk, 0, stream>>>(Atxt, Wbf + (size_t)6 * kDM * kDM, bv_t, yv + (size_t)kLI * kDM,  kLT, kDM, kDM);

    // 3) RMS + RoPE + pack q/k; pack v transposed
    rms_rope_pack_kernel<<<kL, 256, 0, stream>>>(yq, qn, qn_t, rope, mask, 0, qbf);
    rms_rope_pack_kernel<<<kL, 256, 0, stream>>>(yk, kn, kn_t, rope, mask, 1, kbf);
    {
        size_t n = (size_t)kL * kDM;
        v_pack_kernel<<<(unsigned)((n + 255) / 256), 256, 0, stream>>>(yv, vtbf);
    }

    // 4) Flash attention
    dim3 agrid(kL / 128, kH);             // (19, 20)
    flash_attn_kernel<<<agrid, blk, 0, stream>>>(qbf, kbf, vtbf, attn);

    // 5) Output projections straight into d_out (img block then txt block)
    float* out_img = (float*)d_out;
    float* out_txt = out_img + (size_t)kLI * kDM;
    gemm_bf16_kernel<<<gridImg, blk, 0, stream>>>(attn,                       Wbf + (size_t)3 * kDM * kDM, bo,   out_img, kLI, kDM, kDM);
    gemm_bf16_kernel<<<gridTxt, blk, 0, stream>>>(attn + (size_t)kLI * kDM,  Wbf + (size_t)7 * kDM * kDM, bo_t, out_txt, kLT, kDM, kDM);
}
